// SelfAttention_52828097741429
// MI455X (gfx1250) — matrix operand order repaired and checked by reading, not compiled, not run on hardware
//
#include <hip/hip_runtime.h>
#include <hip/hip_bf16.h>
#include <stdint.h>

typedef __bf16 v16bf __attribute__((ext_vector_type(16)));
typedef float  v8f   __attribute__((ext_vector_type(8)));
typedef unsigned int u32x4 __attribute__((ext_vector_type(4)));
typedef int          i32x8 __attribute__((ext_vector_type(8)));
typedef int          i32x4 __attribute__((ext_vector_type(4)));

union FragB16 { v16bf v; uint4 q[2]; };

__device__ __forceinline__ uint16_t f2bf(float f) {
  union { float f; uint32_t u; } c; c.f = f;
  uint32_t u = c.u;
  u += 0x7FFFu + ((u >> 16) & 1u);   // round-to-nearest-even
  return (uint16_t)(u >> 16);
}

// lane-xor butterfly within 16-lane groups via DS_SWIZZLE (group-of-32 mode,
// and_mask=0x1f, or_mask=0, xor_mask=m) -- avoids shfl's bounds-clamp VALU spam
#define SWZ_XOR(x, m) \
  __int_as_float(__builtin_amdgcn_ds_swizzle(__float_as_int(x), 0x7C00 | (m)))

// ---------------------------------------------------------------------------
// TDM: DMA a 2-D bf16 tile (tile_d1 rows x tile_d0 elems, row stride in elems)
// from global into LDS at lds_addr, inserting LDS padding (pad codes per D#).
// Issued by whichever wave executes it; completion tracked on TENSORcnt.
// ---------------------------------------------------------------------------
__device__ __forceinline__ void tdm_load_2d(uint32_t lds_addr, const void* gptr,
                                            uint32_t tile_d0, uint32_t tile_d1,
                                            uint64_t row_stride,
                                            uint32_t pad_interval_code,
                                            uint32_t pad_amount_code)
{
  const uint64_t ga = (uint64_t)gptr;
  u32x4 g0;
  g0[0] = 1u;                                            // count=1, user mode
  g0[1] = lds_addr;                                      // LDS byte address
  g0[2] = (uint32_t)ga;                                  // global_addr[31:0]
  g0[3] = (uint32_t)((ga >> 32) & 0x1FFFFFFu) | (2u << 30);  // [56:32] | type=2

  const uint32_t dim0 = 0x40000000u;                     // huge: never clip
  const uint32_t dim1 = 0x40000000u;
  i32x8 g1;
  g1[0] = (int)((1u << 16)                               // data_size = 2 bytes
              | (1u << 20)                               // pad_enable
              | (pad_interval_code << 22)
              | (pad_amount_code << 25));
  g1[1] = (int)((dim0 & 0xFFFFu) << 16);                 // tensor_dim0 lo
  g1[2] = (int)((dim0 >> 16) | ((dim1 & 0xFFFFu) << 16));
  g1[3] = (int)((dim1 >> 16) | (tile_d0 << 16));         // tile_dim0
  g1[4] = (int)(tile_d1 & 0xFFFFu);                      // tile_dim1, tile_dim2=0
  g1[5] = (int)(uint32_t)row_stride;                     // dim0_stride[31:0]
  g1[6] = (int)(uint32_t)((row_stride >> 32) & 0xFFFFu); // dim0_stride[47:32]
  g1[7] = 0;

  i32x4 z4; z4[0] = z4[1] = z4[2] = z4[3] = 0;
#if __clang_major__ >= 23
  i32x8 z8; for (int i = 0; i < 8; ++i) z8[i] = 0;
  __builtin_amdgcn_tensor_load_to_lds(g0, g1, z4, z4, z8, 0);
#else
  __builtin_amdgcn_tensor_load_to_lds(g0, g1, z4, z4, 0);
#endif
}

// ---------------------------------------------------------------------------
// Tiled transpose + fp32 -> bf16 convert:  WT[k][j] = bf16(W[j][k])
// ---------------------------------------------------------------------------
__global__ __launch_bounds__(256)
void transpose_cvt(const float* __restrict__ W, uint16_t* __restrict__ WT,
                   int R, int Ccol)
{
  __shared__ float tile[32][33];
  const int bx = blockIdx.x * 32;            // k
  const int by = blockIdx.y * 32;            // j
  const int tx = threadIdx.x & 31;
  const int ty = threadIdx.x >> 5;
  #pragma unroll
  for (int i = ty; i < 32; i += 8)
    tile[i][tx] = W[(size_t)(by + i) * Ccol + bx + tx];
  __syncthreads();
  #pragma unroll
  for (int i = ty; i < 32; i += 8)
    WT[(size_t)(bx + i) * R + by + tx] = f2bf(tile[tx][i]);
}

// ---------------------------------------------------------------------------
// bf16 WMMA GEMM:  C[m][n] = sum_k A[m][k] * Bt[k][n] + bias[n]
// B tile (and bf16 A tile) staged by the Tensor Data Mover; fp32 A converted
// through VGPRs. 128x128 tile, 8 waves, K-step 32, double-buffered LDS.
// ---------------------------------------------------------------------------
template<bool A_F32, bool OUT_F32>
__global__ __launch_bounds__(256)
void gemm_bf16(const void* __restrict__ Aptr,
               const uint16_t* __restrict__ Bt,
               const float* __restrict__ bias,
               void* __restrict__ Cptr,
               int M, int N, int K)
{
  __shared__ uint16_t As[2][128][40];   // 128 m-rows x 32 k  (pad 8)
  __shared__ uint16_t Bs[2][32][136];   // 32 k-rows x 128 n  (pad 8)

  const int tid  = threadIdx.x;
  const int lane = tid & 31;
  const int wave = tid >> 5;
  const int m0 = blockIdx.y * 128;
  const int n0 = blockIdx.x * 128;

  const int rowbase = (wave & 3) * 32;  // 4 waves along M
  const int colbase = (wave >> 2) * 64; // 2 waves along N

  v8f acc[2][4];
  #pragma unroll
  for (int i = 0; i < 2; ++i)
    #pragma unroll
    for (int j = 0; j < 4; ++j)
      #pragma unroll
      for (int e = 0; e < 8; ++e)
        acc[i][j][e] = 0.0f;

  auto issue_loads = [&](int buf, int kblk) {
    if (wave == 0) {
      // B tile: 32 rows of 128 bf16 from Bt[kblk..][n0..]; LDS stride 136
      tdm_load_2d((uint32_t)(uintptr_t)&Bs[buf][0][0],
                  Bt + (size_t)kblk * N + n0,
                  128, 32, (uint64_t)N, /*interval 64dw*/5, /*pad 4dw*/3);
      if (!A_F32) {
        // A tile: 128 rows of 32 bf16; LDS stride 40
        tdm_load_2d((uint32_t)(uintptr_t)&As[buf][0][0],
                    (const uint16_t*)Aptr + (size_t)m0 * K + kblk,
                    32, 128, (uint64_t)K, /*interval 16dw*/3, /*pad 4dw*/3);
      }
    }
    if (A_F32) {   // fp32 -> bf16 conversion path (all 256 threads)
      const int r   = tid >> 1;
      const int kc2 = (tid & 1) * 16;
      const float* Arow = (const float*)Aptr + (size_t)(m0 + r) * K + kblk + kc2;
      #pragma unroll
      for (int c = 0; c < 4; ++c) {
        float4 f = *(const float4*)(Arow + c * 4);
        uint32_t p0 = f2bf(f.x) | ((uint32_t)f2bf(f.y) << 16);
        uint32_t p1 = f2bf(f.z) | ((uint32_t)f2bf(f.w) << 16);
        uint32_t* d = (uint32_t*)&As[buf][r][kc2 + c * 4];
        d[0] = p0; d[1] = p1;
      }
    }
  };

  issue_loads(0, 0);
  if (wave == 0) __builtin_amdgcn_s_wait_tensorcnt(0);
  __syncthreads();

  const int nk   = K >> 5;
  const int arow = lane & 15;
  const int kc   = (lane >> 4) << 3;           // 0 or 8

  for (int kt = 0; kt < nk; ++kt) {
    const int cur = kt & 1;
    if (kt + 1 < nk) issue_loads(cur ^ 1, (kt + 1) << 5);

    FragB16 af[2], bfr[4];
    #pragma unroll
    for (int i = 0; i < 2; ++i) {
      const uint16_t* p = &As[cur][rowbase + i * 16 + arow][kc];
      af[i].q[0] = *(const uint4*)p;          // K = kc .. kc+7
      af[i].q[1] = *(const uint4*)(p + 16);   // K = kc+16 .. kc+23
    }
    #pragma unroll
    for (int j = 0; j < 4; ++j) {
      const uint16_t* p = &Bs[cur][kc][colbase + j * 16 + arow];  // lane = n, elems = k
      uint16_t* d = (uint16_t*)&bfr[j];
      #pragma unroll
      for (int s = 0; s < 8; ++s) { d[s] = p[s * 136]; d[8 + s] = p[(16 + s) * 136]; }
    }
    #pragma unroll
    for (int i = 0; i < 2; ++i)
      #pragma unroll
      for (int j = 0; j < 4; ++j)
        acc[i][j] = __builtin_amdgcn_wmma_f32_16x16x32_bf16(
            false, af[i].v, false, bfr[j].v, (short)0, acc[i][j], false, false);

    if (wave == 0) __builtin_amdgcn_s_wait_tensorcnt(0);
    __syncthreads();
  }

  // epilogue: bias add + store
  const int hi = lane >> 4;
  const int ln = lane & 15;
  #pragma unroll
  for (int j = 0; j < 4; ++j) {
    const int col = n0 + colbase + j * 16 + ln;
    const float bv = bias[col];
    #pragma unroll
    for (int i = 0; i < 2; ++i) {
      #pragma unroll
      for (int r = 0; r < 8; ++r) {
        const int row = m0 + rowbase + i * 16 + hi * 8 + r;
        const float v = acc[i][j][r] + bv;
        if (OUT_F32) ((float*)Cptr)[(size_t)row * N + col] = v;
        else ((uint16_t*)Cptr)[(size_t)row * N + col] = f2bf(v);
      }
    }
  }
}

// ---------------------------------------------------------------------------
// Per-token head-attention. One wave per token (H=16, DH=128).
// S = q k^T / sqrt(C) (16x16, K=128) -> softmax rows -> O = P v (16x128)
// Store scrambled: final[b, h*128 + n/16, (n%16)*128 + d] = O[h][d]   (bf16)
// ---------------------------------------------------------------------------
__global__ __launch_bounds__(256)
void attention_kernel(const uint16_t* __restrict__ qkv,
                      uint16_t* __restrict__ finalb)
{
  __shared__ uint16_t kT[8][128][24];   // per-wave transposed k: kT[kk][g]
  __shared__ uint16_t Ps[8][16][40];    // per-wave P patch, cols 16..31 zero

  const int tid  = threadIdx.x;
  const int lane = tid & 31;
  const int w    = tid >> 5;
  const int m    = blockIdx.x * 8 + w;  // token
  const int b    = m >> 11;
  const int n    = m & 2047;

  const uint16_t* row = qkv + (size_t)m * 6144;  // [q | k | v]

  // ---- stage k transposed into LDS ----
  {
    const int g  = lane >> 1;
    const int kb = (lane & 1) * 64;
    const uint16_t* kp = row + 2048 + g * 128 + kb;
    #pragma unroll
    for (int c = 0; c < 8; ++c) {
      uint4 q = *(const uint4*)(kp + c * 8);
      const uint16_t* e = (const uint16_t*)&q;
      #pragma unroll
      for (int t = 0; t < 8; ++t)
        kT[w][kb + c * 8 + t][g] = e[t];
    }
    uint4 z; z.x = z.y = z.z = z.w = 0;       // zero the K=16..31 pad of Ps
    *(uint4*)&Ps[w][lane & 15][16 + (lane >> 4) * 8] = z;
  }
  asm volatile("s_wait_dscnt 0x0" ::: "memory");

  // ---- S = q k^T : 4 WMMA k-steps of 32 ----
  const int hrow = lane & 15;
  const int kc   = (lane >> 4) << 3;
  v8f s;
  #pragma unroll
  for (int e = 0; e < 8; ++e) s[e] = 0.0f;

  #pragma unroll
  for (int st = 0; st < 4; ++st) {
    FragB16 aq, bk;
    const uint16_t* qp = row + hrow * 128 + st * 32 + kc;
    aq.q[0] = *(const uint4*)qp;
    aq.q[1] = *(const uint4*)(qp + 16);
    const uint16_t* kp = row + 2048 + hrow * 128 + st * 32 + kc;  // lane = g
    bk.q[0] = *(const uint4*)kp;
    bk.q[1] = *(const uint4*)(kp + 16);
    s = __builtin_amdgcn_wmma_f32_16x16x32_bf16(
        false, aq.v, false, bk.v, (short)0, s, false, false);
  }

  // ---- softmax over g (16-lane groups) via ds_swizzle butterflies ----
  const float scale = 0.02209708691f;  // 1/sqrt(2048)
  #pragma unroll
  for (int r = 0; r < 8; ++r) {
    float v = s[r] * scale;
    float mx = v;
    mx = fmaxf(mx, SWZ_XOR(mx, 1));
    mx = fmaxf(mx, SWZ_XOR(mx, 2));
    mx = fmaxf(mx, SWZ_XOR(mx, 4));
    mx = fmaxf(mx, SWZ_XOR(mx, 8));
    float e = __expf(v - mx);
    float sm = e;
    sm += SWZ_XOR(sm, 1);
    sm += SWZ_XOR(sm, 2);
    sm += SWZ_XOR(sm, 4);
    sm += SWZ_XOR(sm, 8);
    Ps[w][(lane >> 4) * 8 + r][lane & 15] = f2bf(e / sm);
  }
  asm volatile("s_wait_dscnt 0x0" ::: "memory");

  // ---- P as 16x32 A fragment (upper K half zero) ----
  FragB16 ap;
  {
    const uint16_t* pp = &Ps[w][lane & 15][kc];
    ap.q[0] = *(const uint4*)pp;
    ap.q[1] = *(const uint4*)(pp + 16);     // zeros
  }

  // ---- O = P v, 8 d-tiles of 16; store scrambled bf16 ----
  const uint16_t* vbase = row + 4096 + (lane & 15);
  #pragma unroll
  for (int j = 0; j < 8; ++j) {
    FragB16 bv;
    {
      uint16_t* d = (uint16_t*)&bv;
      #pragma unroll
      for (int s = 0; s < 8; ++s) { d[s] = vbase[(kc + s) * 128 + j * 16]; d[8 + s] = 0; }
    }
    v8f o;
    #pragma unroll
    for (int e = 0; e < 8; ++e) o[e] = 0.0f;
    o = __builtin_amdgcn_wmma_f32_16x16x32_bf16(
        false, ap.v, false, bv.v, (short)0, o, false, false);

    const int col = (n & 15) * 128 + j * 16 + (lane & 15);
    #pragma unroll
    for (int r = 0; r < 8; ++r) {
      const int h    = r + (lane >> 4) * 8;
      const int grow = b * 2048 + h * 128 + (n >> 4);
      finalb[(size_t)grow * 2048 + col] = f2bf(o[r]);
    }
  }
}

// ---------------------------------------------------------------------------
extern "C" void kernel_launch(void* const* d_in, const int* in_sizes, int n_in,
                              void* d_out, int out_size, void* d_ws, size_t ws_size,
                              hipStream_t stream)
{
  const float* x     = (const float*)d_in[0];  // 4 x 2048 x 2048
  const float* w_qkv = (const float*)d_in[1];  // 6144 x 2048
  const float* b_qkv = (const float*)d_in[2];  // 6144
  const float* w_out = (const float*)d_in[3];  // 2048 x 2048
  const float* b_out = (const float*)d_in[4];  // 2048
  float* out = (float*)d_out;                  // 8192 x 2048

  // workspace layout (bf16), total 160 MiB
  uint16_t* wqkvT  = (uint16_t*)d_ws;                    // 2048 x 6144
  uint16_t* woutT  = wqkvT + (size_t)2048 * 6144;        // 2048 x 2048
  uint16_t* qkv    = woutT + (size_t)2048 * 2048;        // 8192 x 6144
  uint16_t* finalb = qkv   + (size_t)8192 * 6144;        // 8192 x 2048

  dim3 blk(256);
  transpose_cvt<<<dim3(2048 / 32, 6144 / 32), blk, 0, stream>>>(w_qkv, wqkvT, 6144, 2048);
  transpose_cvt<<<dim3(2048 / 32, 2048 / 32), blk, 0, stream>>>(w_out, woutT, 2048, 2048);

  gemm_bf16<true, false><<<dim3(6144 / 128, 8192 / 128), blk, 0, stream>>>(
      x, wqkvT, b_qkv, qkv, 8192, 6144, 2048);

  attention_kernel<<<dim3(8192 / 8), blk, 0, stream>>>(qkv, finalb);

  gemm_bf16<false, true><<<dim3(2048 / 128, 8192 / 128), blk, 0, stream>>>(
      finalb, woutT, b_out, out, 8192, 2048, 2048);
}
